// MoE_78932908966071
// MI455X (gfx1250) — compile-verified, matrix operand order checked by reading, not run on hardware
//
#include <hip/hip_runtime.h>
#include <math.h>

// Problem constants (B=2, S=2048 -> N_TOK=4096), D=1024, F=4096, E=8, top-k=2
#define N_TOK 4096
#define DIM   1024
#define FF    4096
#define NE    8

#define TM    32     // tokens per block tile
#define TF    64     // F-chunk width (phase-1 N, phase-2 K)
#define KC    128    // D-chunk depth for phase-1 K loop
#define DC    128    // D-chunk width for phase-2 output staging
#define LDSW  136    // padded row stride for KC-wide rows (272B, 16B aligned)
#define LDSN  72     // padded row stride for TF-wide rows  (144B, 16B aligned)

typedef __attribute__((ext_vector_type(16))) __bf16 bf16x16;
typedef __attribute__((ext_vector_type(8)))  __bf16 bf16x8;
typedef __attribute__((ext_vector_type(4)))  __bf16 bf16x4;
typedef __attribute__((ext_vector_type(8)))  float  f32x8;
typedef __attribute__((ext_vector_type(4)))  float  f32x4;

union BF16Frag { bf16x16 v; bf16x8 h[2]; };

// ---------------------------------------------------------------------------
// Kernel 0: zero the output accumulator and the per-expert counters.
// ---------------------------------------------------------------------------
__global__ void moe_init(float* __restrict__ out, int* __restrict__ cnt, int total4) {
  int idx = blockIdx.x * blockDim.x + threadIdx.x;
  if (idx < total4) {
    float4 z; z.x = 0.f; z.y = 0.f; z.z = 0.f; z.w = 0.f;
    reinterpret_cast<float4*>(out)[idx] = z;
  }
  if (blockIdx.x == 0 && threadIdx.x < NE) cnt[threadIdx.x] = 0;
}

// ---------------------------------------------------------------------------
// Kernel 1: router. One wave32 per token: scores = x . Wr[e], top-2, build
// per-expert token lists (jax top_k tie-break: lower index wins -> strict >).
// ---------------------------------------------------------------------------
__global__ void moe_router(const float* __restrict__ x, const float* __restrict__ Wr,
                           int* __restrict__ cnt, int* __restrict__ list) {
  const int wave = threadIdx.x >> 5;
  const int lane = threadIdx.x & 31;
  const int t = blockIdx.x * 8 + wave;

  float acc[NE];
#pragma unroll
  for (int e = 0; e < NE; ++e) acc[e] = 0.f;

  const float* xr = x + (size_t)t * DIM;
  for (int d = lane; d < DIM; d += 32) {
    float xv = xr[d];
#pragma unroll
    for (int e = 0; e < NE; ++e) acc[e] += xv * Wr[e * DIM + d];
  }
#pragma unroll
  for (int e = 0; e < NE; ++e) {
#pragma unroll
    for (int off = 16; off > 0; off >>= 1) acc[e] += __shfl_xor(acc[e], off, 32);
  }
  if (lane == 0) {
    int b1 = 0;
#pragma unroll
    for (int e = 1; e < NE; ++e) if (acc[e] > acc[b1]) b1 = e;
    int b2 = (b1 == 0) ? 1 : 0;
#pragma unroll
    for (int e = 0; e < NE; ++e) if (e != b1 && acc[e] > acc[b2]) b2 = e;
    int p1 = atomicAdd(&cnt[b1], 1); list[b1 * N_TOK + p1] = t;
    int p2 = atomicAdd(&cnt[b2], 1); list[b2 * N_TOK + p2] = t;
  }
}

// ---------------------------------------------------------------------------
// Kernel 2: expert FFN with bf16 WMMA (f32 accumulate).
// Block = 256 threads (8 wave32), tile = 32 tokens of one expert.
//   phase 1: Hc[32,TF] = gelu(X[32,1024] @ W1[:, fc:fc+TF] + b1)   (per fc)
//            K staged in KC=128 chunks -> 4 WMMA per barrier pair.
//   phase 2: Y[32,1024] += Hc @ W2[fc:fc+TF, :]   (8 DC=128 chunks,
//            each wave owns 2 column tiles -> 4 WMMA per barrier pair).
//   epilogue: out[t] += 0.5*(Y + b2) via atomicAdd (exactly 2 adds/elem ->
//             bitwise deterministic since fp add is commutative).
// Weight transposes are done as in-register 4x4 blocks with packed b64
// LDS stores; X staging uses packed b128 stores.
// ---------------------------------------------------------------------------
__global__ void __launch_bounds__(256)
moe_ffn(const float* __restrict__ x,
        const float* __restrict__ W1, const float* __restrict__ b1,
        const float* __restrict__ W2, const float* __restrict__ b2,
        const int* __restrict__ cnt, const int* __restrict__ list,
        float* __restrict__ out) {
  const int e    = blockIdx.y;
  const int tile = blockIdx.x;
  const int ce   = cnt[e];
  if (tile * TM >= ce) return;

  __shared__ __align__(16) __bf16 Xs [TM][LDSW];   // [tok][k]   8.7 KB
  __shared__ __align__(16) __bf16 W1s[TF][LDSW];   // [f][k]    17.4 KB
  __shared__ __align__(16) __bf16 Hs [TM][LDSN];   // [tok][f]   4.6 KB
  __shared__ __align__(16) __bf16 W2s[DC][LDSN];   // [d][k]    18.4 KB
  __shared__ int tokLoad[TM];
  __shared__ int tokStore[TM];

  const int tid  = threadIdx.x;
  const int wave = tid >> 5;
  const int lane = tid & 31;
  const int mi   = wave >> 2;   // token row-tile (0..1)
  const int fi   = wave & 3;    // column tile group (0..3)
  const int lrow = lane & 15;   // M (A) / N (B) index for this lane
  const int lhi  = lane >> 4;   // K-half select per ISA 16-bit layout

  if (tid < TM) {
    int r  = tile * TM + tid;
    int tk = (r < ce) ? list[e * N_TOK + r] : -1;
    tokStore[tid] = tk;
    tokLoad[tid]  = (tk < 0) ? list[e * N_TOK + tile * TM] : tk;
  }
  __syncthreads();

  f32x8 yacc[16];  // [dcx*2 + sub]
#pragma unroll
  for (int i = 0; i < 16; ++i) yacc[i] = {};

  const float* W1e = W1 + (size_t)e * DIM * FF;
  const float* W2e = W2 + (size_t)e * FF * DIM;

  for (int fc = 0; fc < FF; fc += TF) {
    // ----------------- phase 1: Hc = gelu(X @ W1 + b1) -----------------
    f32x8 hc = {};
    for (int kc = 0; kc < DIM; kc += KC) {
      { // stage X tile (32 x KC) fp32 -> bf16 row-major, packed b128 stores
        int row = tid >> 3;
        int kk  = (tid & 7) * 16;
        const float* src = x + (size_t)tokLoad[row] * DIM + kc + kk;
        f32x4 a = reinterpret_cast<const f32x4*>(src)[0];
        f32x4 b = reinterpret_cast<const f32x4*>(src)[1];
        f32x4 c = reinterpret_cast<const f32x4*>(src)[2];
        f32x4 d = reinterpret_cast<const f32x4*>(src)[3];
        bf16x8 o0, o1;
#pragma unroll
        for (int j = 0; j < 4; ++j) {
          o0[j]     = (__bf16)a[j];
          o0[j + 4] = (__bf16)b[j];
          o1[j]     = (__bf16)c[j];
          o1[j + 4] = (__bf16)d[j];
        }
        *reinterpret_cast<bf16x8*>(&Xs[row][kk])     = o0;
        *reinterpret_cast<bf16x8*>(&Xs[row][kk + 8]) = o1;
      }
      // stage W1 chunk (KC x TF) -> W1s[f][k] via in-register 4x4 transpose
#pragma unroll
      for (int it = 0; it < 2; ++it) {
        int b  = tid + 256 * it;
        int fq = b & 15, kq = b >> 4;
        int f0 = fq * 4, k0 = kq * 4;
        const float* src = W1e + (size_t)(kc + k0) * FF + fc + f0;
        f32x4 r0 = reinterpret_cast<const f32x4*>(src)[0];
        f32x4 r1 = reinterpret_cast<const f32x4*>(src + FF)[0];
        f32x4 r2 = reinterpret_cast<const f32x4*>(src + 2 * FF)[0];
        f32x4 r3 = reinterpret_cast<const f32x4*>(src + 3 * FF)[0];
        if (kc + KC < DIM) __builtin_prefetch(src + (size_t)KC * FF, 0, 0);
#pragma unroll
        for (int jf = 0; jf < 4; ++jf) {
          bf16x4 p = { (__bf16)r0[jf], (__bf16)r1[jf], (__bf16)r2[jf], (__bf16)r3[jf] };
          *reinterpret_cast<bf16x4*>(&W1s[f0 + jf][k0]) = p;
        }
      }
      __syncthreads();
#pragma unroll
      for (int ks = 0; ks < KC; ks += 32) {
        BF16Frag a, b;
        const __bf16* pa = &Xs[mi * 16 + lrow][ks + lhi * 8];
        a.h[0] = *reinterpret_cast<const bf16x8*>(pa);
        a.h[1] = *reinterpret_cast<const bf16x8*>(pa + 16);
        const __bf16* pb = &W1s[fi * 16 + lrow][ks + lhi * 8];
        b.h[0] = *reinterpret_cast<const bf16x8*>(pb);
        b.h[1] = *reinterpret_cast<const bf16x8*>(pb + 16);
        hc = __builtin_amdgcn_wmma_f32_16x16x32_bf16(false, a.v, false, b.v,
                                                     (short)0, hc, false, false);
      }
      __syncthreads();
    }
    { // bias + exact gelu, write Hc -> Hs (bf16); C layout: (m = i+8*lhi, n = lrow)
      float bb = b1[e * FF + fc + fi * 16 + lrow];
#pragma unroll
      for (int i = 0; i < 8; ++i) {
        float v = hc[i] + bb;
        float g = 0.5f * v * (1.0f + erff(v * 0.70710678118654752f));
        Hs[mi * 16 + i + lhi * 8][fi * 16 + lrow] = (__bf16)g;
      }
    }
    __syncthreads();
    // ----------------- phase 2: Y += Hc @ W2 -----------------
    for (int dcx = 0; dcx < 8; ++dcx) {
      int dc = dcx * DC;
      // stage W2 chunk (TF x DC) -> W2s[d][k] via in-register 4x4 transpose
#pragma unroll
      for (int it = 0; it < 2; ++it) {
        int b  = tid + 256 * it;
        int dq = b & 31, kq = b >> 5;
        int d0 = dq * 4, k0 = kq * 4;
        const float* src = W2e + (size_t)(fc + k0) * DIM + dc + d0;
        f32x4 r0 = reinterpret_cast<const f32x4*>(src)[0];
        f32x4 r1 = reinterpret_cast<const f32x4*>(src + DIM)[0];
        f32x4 r2 = reinterpret_cast<const f32x4*>(src + 2 * DIM)[0];
        f32x4 r3 = reinterpret_cast<const f32x4*>(src + 3 * DIM)[0];
        if (dcx < 7) __builtin_prefetch(src + DC, 0, 0);
#pragma unroll
        for (int jd = 0; jd < 4; ++jd) {
          bf16x4 p = { (__bf16)r0[jd], (__bf16)r1[jd], (__bf16)r2[jd], (__bf16)r3[jd] };
          *reinterpret_cast<bf16x4*>(&W2s[d0 + jd][k0]) = p;
        }
      }
      __syncthreads();
#pragma unroll
      for (int sub = 0; sub < 2; ++sub) {
        const int di = fi * 2 + sub;
#pragma unroll
        for (int ks = 0; ks < TF; ks += 32) {
          BF16Frag a, b;
          const __bf16* pa = &Hs[mi * 16 + lrow][ks + lhi * 8];
          a.h[0] = *reinterpret_cast<const bf16x8*>(pa);
          a.h[1] = *reinterpret_cast<const bf16x8*>(pa + 16);
          const __bf16* pb = &W2s[di * 16 + lrow][ks + lhi * 8];
          b.h[0] = *reinterpret_cast<const bf16x8*>(pb);
          b.h[1] = *reinterpret_cast<const bf16x8*>(pb + 16);
          yacc[dcx * 2 + sub] =
              __builtin_amdgcn_wmma_f32_16x16x32_bf16(false, a.v, false, b.v,
                                                      (short)0, yacc[dcx * 2 + sub],
                                                      false, false);
        }
      }
      __syncthreads();
    }
  }

  // ----------------- epilogue: out[t] += 0.5*(y + b2) -----------------
#pragma unroll
  for (int dcx = 0; dcx < 8; ++dcx) {
#pragma unroll
    for (int sub = 0; sub < 2; ++sub) {
      int col = dcx * DC + (fi * 2 + sub) * 16 + lrow;
      float bb = b2[e * DIM + col];
#pragma unroll
      for (int i = 0; i < 8; ++i) {
        int row = mi * 16 + i + lhi * 8;
        int t = tokStore[row];
        if (t >= 0)
          atomicAdd(&out[(size_t)t * DIM + col], 0.5f * (yacc[dcx * 2 + sub][i] + bb));
      }
    }
  }
}

// ---------------------------------------------------------------------------
// Launch wrapper. Inputs: x, Wr, W1, b1, W2, b2 (all fp32). Output fp32 [N,D].
// Workspace: cnt (8 ints, padded) + list (E*N ints) ~= 128 KB.
// ---------------------------------------------------------------------------
extern "C" void kernel_launch(void* const* d_in, const int* in_sizes, int n_in,
                              void* d_out, int out_size, void* d_ws, size_t ws_size,
                              hipStream_t stream) {
  const float* x  = (const float*)d_in[0];
  const float* Wr = (const float*)d_in[1];
  const float* W1 = (const float*)d_in[2];
  const float* b1 = (const float*)d_in[3];
  const float* W2 = (const float*)d_in[4];
  const float* b2 = (const float*)d_in[5];
  float* out = (float*)d_out;

  int* cnt  = (int*)d_ws;
  int* list = cnt + 32;  // keep list 128B aligned

  const int total4 = (N_TOK * DIM) / 4;
  moe_init<<<(total4 + 255) / 256, 256, 0, stream>>>(out, cnt, total4);
  moe_router<<<N_TOK / 8, 256, 0, stream>>>(x, Wr, cnt, list);
  dim3 grid(N_TOK / TM, NE);
  moe_ffn<<<grid, 256, 0, stream>>>(x, W1, b1, W2, b2, cnt, list, out);
}
